// LSTMModel_79559974191164
// MI455X (gfx1250) — compile-verified, matrix operand order checked by reading, not run on hardware
//
#include <hip/hip_runtime.h>

typedef __attribute__((ext_vector_type(4)))  __bf16 v4bf;
typedef __attribute__((ext_vector_type(8)))  __bf16 v8bf;
typedef __attribute__((ext_vector_type(16))) __bf16 v16bf;
typedef __attribute__((ext_vector_type(8)))  float  v8f;

#define B_   256
#define T_   512
#define I_   64
#define H_   512
#define O_   64
#define KTOT 576              // H + I (concatenated [h | x_t] K-dim)
#define NWG  32               // persistent workgroups
#define UPW  16               // hidden units per WG
#define NCOL 64               // 4 gates * UPW columns per WG
#define KP   584              // padded LDS K-stride in bf16 elems (mult of 8)

#define SMEM_BYTES (NCOL * KP * 2 + NCOL * 4)

// ---- workspace layout (bytes) ----
#define WS_CNT  0
#define WS_H0   1024
#define WS_H1   (WS_H0 + B_ * H_ * 2)          // 263,168
#define WS_C    (WS_H1 + B_ * H_ * 2)          // 525,312
#define WS_XBF  (WS_C  + B_ * H_ * 4)          // 1,049,600
// total ~17.8 MB

static __device__ __forceinline__ float fsig(float x) {
    return __builtin_amdgcn_rcpf(1.0f + __expf(-x));
}
static __device__ __forceinline__ float ftanh(float x) {
    float e = __expf(2.0f * x);
    return 1.0f - 2.0f * __builtin_amdgcn_rcpf(e + 1.0f);
}

// Load one 16xK bf16 fragment half-pair: elems [0..7] = K kb..kb+7, [8..15] = K kb+16..kb+23
template <typename P>
static __device__ __forceinline__ v16bf ld_frag(P p) {
    v8bf lo = *(const v8bf*)p;
    v8bf hi = *(const v8bf*)(p + 16);
    return __builtin_shufflevector(lo, hi, 0, 1, 2, 3, 4, 5, 6, 7,
                                           8, 9, 10, 11, 12, 13, 14, 15);
}

static __device__ __forceinline__ v8f wmma_bf16(v16bf a, v16bf b, v8f c) {
    return __builtin_amdgcn_wmma_f32_16x16x32_bf16(false, a, false, b,
                                                   (short)0, c, false, false);
}

// ---------------- prologue: convert x to bf16, zero h0/h1/c, reset counter ----------------
__global__ void __launch_bounds__(256) lstm_init(const float* __restrict__ x,
                                                 __bf16* __restrict__ xbf,
                                                 __bf16* __restrict__ h0,
                                                 __bf16* __restrict__ h1,
                                                 float* __restrict__ cbuf,
                                                 unsigned* __restrict__ cnt) {
    int id = blockIdx.x * 256 + threadIdx.x;       // 0 .. 2,097,151 (x elems / 4)
    float4 v = ((const float4*)x)[id];
    v4bf o;
    o[0] = (__bf16)v.x; o[1] = (__bf16)v.y; o[2] = (__bf16)v.z; o[3] = (__bf16)v.w;
    *(v4bf*)(xbf + 4 * id) = o;
    if (id < (B_ * H_) / 4) {
        v4bf z; z[0] = z[1] = z[2] = z[3] = (__bf16)0.0f;
        *(v4bf*)(h0 + 4 * id) = z;
        *(v4bf*)(h1 + 4 * id) = z;
        float4 zf = {0.f, 0.f, 0.f, 0.f};
        ((float4*)cbuf)[id] = zf;
    }
    if (id == 0) *cnt = 0u;
}

// ---------------- persistent LSTM recurrence ----------------
__global__ void __launch_bounds__(256, 1)
lstm_persist(const float* __restrict__ W_ih, const float* __restrict__ W_hh,
             const float* __restrict__ b_ih, const float* __restrict__ b_hh,
             __bf16* __restrict__ h0, __bf16* __restrict__ h1,
             float* __restrict__ cbuf, const __bf16* __restrict__ xbf,
             unsigned* __restrict__ cnt) {
    extern __shared__ __bf16 smemRaw[];
    __bf16* sW   = smemRaw;                          // [NCOL][KP] bf16
    float*  sBias = (float*)(smemRaw + NCOL * KP);   // [NCOL]

    const int tid = threadIdx.x;
    const int wg  = blockIdx.x;

    // Stage [W_hh | W_ih] slice (bf16) + bias into LDS, once.
    for (int e = tid; e < NCOL * KTOT; e += 256) {
        int n = e / KTOT, k = e - n * KTOT;
        int q = n >> 4;                       // gate (0=i,1=f,2=g,3=o)
        int u = (wg << 4) + (n & 15);         // hidden unit
        int grow = q * H_ + u;                // row in 4H weight matrices
        float w = (k < H_) ? W_hh[grow * H_ + k] : W_ih[grow * I_ + (k - H_)];
        sW[n * KP + k] = (__bf16)w;
    }
    if (tid < NCOL) {
        int q = tid >> 4, u = (wg << 4) + (tid & 15);
        int grow = q * H_ + u;
        sBias[tid] = b_ih[grow] + b_hh[grow];
    }
    __syncthreads();

    const int wv    = tid >> 5;               // wave 0..7 -> M rows [wv*32, wv*32+32)
    const int lane  = tid & 31;
    const int l16   = lane & 15;
    const int khalf = (lane >> 4) << 3;       // 0 or 8
    const int m0    = wv * 32;
    const int rowA0 = m0 + l16;
    const int rowA1 = m0 + 16 + l16;
    const int mhalf = (lane >> 4) << 3;       // C-tile row offset for this lane
    const int hu    = (wg << 4) + l16;        // global hidden unit for this lane

    const float bias_i = sBias[0 + l16];
    const float bias_f = sBias[16 + l16];
    const float bias_g = sBias[32 + l16];
    const float bias_o = sBias[48 + l16];

    for (int t = 0; t < T_; ++t) {
        const __bf16* hRead  = (t & 1) ? h1 : h0;
        __bf16*       hWrite = (t & 1) ? h0 : h1;

        v8f acc[2][4] = {};

        // K over previous h (512)
        const __bf16* a0base = hRead + rowA0 * H_ + khalf;
        const __bf16* a1base = hRead + rowA1 * H_ + khalf;
#pragma unroll 4
        for (int ks = 0; ks < 16; ++ks) {
            const int kb = ks * 32;
            v16bf a0 = ld_frag(a0base + kb);
            v16bf a1 = ld_frag(a1base + kb);
#pragma unroll
            for (int nt = 0; nt < 4; ++nt) {
                v16bf b = ld_frag(sW + (nt * 16 + l16) * KP + kb + khalf);
                acc[0][nt] = wmma_bf16(a0, b, acc[0][nt]);
                acc[1][nt] = wmma_bf16(a1, b, acc[1][nt]);
            }
        }
        // K over x_t (64)
        const __bf16* x0base = xbf + (rowA0 * T_ + t) * I_ + khalf;
        const __bf16* x1base = xbf + (rowA1 * T_ + t) * I_ + khalf;
#pragma unroll
        for (int ks = 0; ks < 2; ++ks) {
            const int kb = ks * 32;
            v16bf a0 = ld_frag(x0base + kb);
            v16bf a1 = ld_frag(x1base + kb);
#pragma unroll
            for (int nt = 0; nt < 4; ++nt) {
                v16bf b = ld_frag(sW + (nt * 16 + l16) * KP + H_ + kb + khalf);
                acc[0][nt] = wmma_bf16(a0, b, acc[0][nt]);
                acc[1][nt] = wmma_bf16(a1, b, acc[1][nt]);
            }
        }

        // Elementwise LSTM update: all 4 gates for (row, hu) live in this lane.
#pragma unroll
        for (int mt = 0; mt < 2; ++mt) {
#pragma unroll
            for (int r = 0; r < 8; ++r) {
                int row = m0 + mt * 16 + mhalf + r;
                float gi = fsig(acc[mt][0][r] + bias_i);
                float gf = fsig(acc[mt][1][r] + bias_f);
                float gg = ftanh(acc[mt][2][r] + bias_g);
                float go = fsig(acc[mt][3][r] + bias_o);
                int idx = row * H_ + hu;
                float c = gf * cbuf[idx] + gi * gg;
                cbuf[idx] = c;
                hWrite[idx] = (__bf16)(go * ftanh(c));
            }
        }

        // Grid-wide barrier: release h writes, wait for all WGs.
        __threadfence();
        __syncthreads();
        if (tid == 0) {
            __hip_atomic_fetch_add(cnt, 1u, __ATOMIC_ACQ_REL, __HIP_MEMORY_SCOPE_AGENT);
            unsigned target = (unsigned)NWG * (unsigned)(t + 1);
            while (__hip_atomic_load(cnt, __ATOMIC_ACQUIRE, __HIP_MEMORY_SCOPE_AGENT) < target) {
                __builtin_amdgcn_s_sleep(2);
            }
        }
        __syncthreads();
        __threadfence();
    }
}

// ---------------- epilogue: out = relu(h_T) @ W_fc^T + b_fc ----------------
__global__ void __launch_bounds__(256) lstm_fc(const __bf16* __restrict__ hT,
                                               const float* __restrict__ W_fc,
                                               const float* __restrict__ b_fc,
                                               float* __restrict__ out) {
    int id = blockIdx.x * 256 + threadIdx.x;   // 16384 = B*O
    int b = id >> 6, o = id & 63;
    const __bf16* hr = hT + b * H_;
    const float*  wr = W_fc + o * H_;
    float acc = 0.0f;
#pragma unroll 8
    for (int k = 0; k < H_; ++k) {
        float h = (float)hr[k];
        h = h > 0.0f ? h : 0.0f;
        acc = fmaf(h, wr[k], acc);
    }
    out[id] = acc + b_fc[o];
}

extern "C" void kernel_launch(void* const* d_in, const int* in_sizes, int n_in,
                              void* d_out, int out_size, void* d_ws, size_t ws_size,
                              hipStream_t stream) {
    const float* x    = (const float*)d_in[0];
    const float* W_ih = (const float*)d_in[1];
    const float* W_hh = (const float*)d_in[2];
    const float* b_ih = (const float*)d_in[3];
    const float* b_hh = (const float*)d_in[4];
    const float* W_fc = (const float*)d_in[5];
    const float* b_fc = (const float*)d_in[6];
    float* out = (float*)d_out;

    char* ws = (char*)d_ws;
    unsigned* cnt  = (unsigned*)(ws + WS_CNT);
    __bf16*   h0   = (__bf16*)(ws + WS_H0);
    __bf16*   h1   = (__bf16*)(ws + WS_H1);
    float*    cbuf = (float*)(ws + WS_C);
    __bf16*   xbf  = (__bf16*)(ws + WS_XBF);

    (void)in_sizes; (void)n_in; (void)out_size; (void)ws_size;

    // allow >64KB dynamic LDS (gfx1250 WGP has 320KB)
    hipFuncSetAttribute((const void*)lstm_persist,
                        hipFuncAttributeMaxDynamicSharedMemorySize, SMEM_BYTES);

    // 1) convert x -> bf16, zero h/c, reset barrier counter
    lstm_init<<<(B_ * T_ * I_ / 4 + 255) / 256, 256, 0, stream>>>(x, xbf, h0, h1, cbuf, cnt);

    // 2) persistent recurrence over all 512 timesteps
    lstm_persist<<<NWG, 256, SMEM_BYTES, stream>>>(W_ih, W_hh, b_ih, b_hh,
                                                   h0, h1, cbuf, xbf, cnt);

    // 3) final FC on h_T (lives in h0 after the last step writes buffer 0)
    lstm_fc<<<(B_ * O_ + 255) / 256, 256, 0, stream>>>(h0, W_fc, b_fc, out);
}
